// SubLSTMCudaCell_84078279786558
// MI455X (gfx1250) — matrix operand order repaired and checked by reading, not compiled, not run on hardware
//
#include <hip/hip_runtime.h>

// ---------------------------------------------------------------------------
// subLSTM cell, fused: gates = sigmoid([old_h|input] @ W^T + b)
//   new_cell = old_cell * f + z - i ; new_h = sigmoid(new_cell) - o
// B=32768, H=I=256 -> K=512, N=4H=1024.
// bf16 WMMA GEMM (v_wmma_f32_16x16x32_bf16) + TDM (tensor_load_to_lds)
// double-buffered A staging + fully fused epilogue.
// ---------------------------------------------------------------------------

typedef __attribute__((ext_vector_type(16))) __bf16         v16bf;
typedef __attribute__((ext_vector_type(8)))  float          v8f;
typedef __attribute__((ext_vector_type(8)))  unsigned short u16x8;
typedef __attribute__((ext_vector_type(16))) unsigned short u16x16;
typedef __attribute__((ext_vector_type(4)))  unsigned int   u32x4;
typedef __attribute__((ext_vector_type(8)))  int            i32x8;
typedef __attribute__((ext_vector_type(4)))  int            i32x4;

static constexpr int Bn  = 32768;  // batch
static constexpr int Hn  = 256;    // hidden
static constexpr int Kn  = 512;    // H + I
static constexpr int LDA = 40;     // LDS row stride, bf16 elems (80B: 64B data + 16B pad)
static constexpr int NSTEP = Kn / 32;

#if defined(__has_builtin)
#  if __has_builtin(__builtin_amdgcn_tensor_load_to_lds) && \
      __has_builtin(__builtin_amdgcn_s_wait_tensorcnt)
#    define SUBLSTM_HAVE_TDM 1
#  endif
#endif
#ifndef SUBLSTM_HAVE_TDM
#  define SUBLSTM_HAVE_TDM 0
#endif

__device__ __forceinline__ unsigned short f32_to_bf16_rne(float f) {
    union { float f; unsigned int u; } v; v.f = f;
    unsigned int u = v.u;
    unsigned int r = u + 0x7FFFu + ((u >> 16) & 1u);
    return (unsigned short)(r >> 16);
}

__device__ __forceinline__ v16bf mk_frag(u16x8 lo, u16x8 hi) {
    u16x16 c = __builtin_shufflevector(lo, hi, 0,1,2,3,4,5,6,7,8,9,10,11,12,13,14,15);
    return __builtin_bit_cast(v16bf, c);
}

__device__ __forceinline__ float sigmoidf(float x) {
    return 1.0f / (1.0f + __expf(-x));
}

#if SUBLSTM_HAVE_TDM
// Issue one TDM load: 2-D bf16 tile 32 (K) x 128 (M) from Xb (row stride 512)
// into LDS at lds_byte_off, padded 4 DWORDs every 16 DWORDs -> 80B LDS rows.
// D# layout per CDNA5 ISA 8.3/8.4: group0 {count,lds,global,type}, group1
// {data_size, pad, tensor dims/strides, tile dims}. Groups 2/3 zero (2-D).
__device__ __forceinline__ void tdm_load_tile_a(const unsigned short* gsrc,
                                                unsigned lds_byte_off) {
    unsigned long long ga = (unsigned long long)gsrc;
    u32x4 g0;
    g0.x = 1u;                                                // count=1, no gather
    g0.y = lds_byte_off;                                      // lds_addr
    g0.z = (unsigned)ga;                                      // global_addr[31:0]
    g0.w = (unsigned)((ga >> 32) & 0x01FFFFFFull) | (2u << 30); // addr[56:32] | type=2
    i32x8 g1;
    g1[0] = (1 << 16)        // data_size = 2 bytes
          | (1 << 20)        // pad_enable
          | (3 << 22)        // pad_interval: every 16 DWORDs (64B)
          | (3 << 25);       // pad_amount: 4 DWORDs (16B)
    g1[1] = (int)(512u << 16);    // tensor_dim0[15:0] = 512 (K extent)
    g1[2] = (int)(32768u << 16);  // tensor_dim0[31:16]=0 | tensor_dim1[15:0]=32768
    g1[3] = (int)(32u << 16);     // tensor_dim1[31:16]=0 | tile_dim0 = 32
    g1[4] = 128;                  // tile_dim1 = 128, tile_dim2 = 0
    g1[5] = 512;                  // tensor_dim0_stride[31:0] = 512 elems
    g1[6] = 0;                    // stride hi | tensor_dim1_stride lo (unused, 2-D)
    g1[7] = 0;
    i32x4 z4 = {0, 0, 0, 0};
#  if defined(__clang_major__) && __clang_major__ >= 23
    i32x8 z8 = {0, 0, 0, 0, 0, 0, 0, 0};
    __builtin_amdgcn_tensor_load_to_lds(g0, g1, z4, z4, z8, 0);
#  else
    __builtin_amdgcn_tensor_load_to_lds(g0, g1, z4, z4, 0);
#  endif
}
#endif

// --- pack X = [old_h | input] -> bf16 [B, 512] row-major ---------------------
__global__ __launch_bounds__(256)
void pack_x_kernel(const float* __restrict__ input,
                   const float* __restrict__ old_h,
                   unsigned short* __restrict__ Xb) {
    long long e = ((long long)blockIdx.x * blockDim.x + threadIdx.x) * 4;
    int b = (int)(e >> 9);
    int j = (int)(e & 511);
    const float* src = (j < 256) ? (old_h + (long long)b * 256 + j)
                                 : (input + (long long)b * 256 + (j - 256));
    float4 f = *(const float4*)src;
    ushort4 o;
    o.x = f32_to_bf16_rne(f.x);
    o.y = f32_to_bf16_rne(f.y);
    o.z = f32_to_bf16_rne(f.z);
    o.w = f32_to_bf16_rne(f.w);
    *(ushort4*)(Xb + e) = o;
}

// --- pack W -> bf16 [1024, 512] ---------------------------------------------
__global__ __launch_bounds__(256)
void pack_w_kernel(const float* __restrict__ W,
                   unsigned short* __restrict__ Wb) {
    long long e = ((long long)blockIdx.x * blockDim.x + threadIdx.x) * 4;
    float4 f = *(const float4*)(W + e);
    ushort4 o;
    o.x = f32_to_bf16_rne(f.x);
    o.y = f32_to_bf16_rne(f.y);
    o.z = f32_to_bf16_rne(f.z);
    o.w = f32_to_bf16_rne(f.w);
    *(ushort4*)(Wb + e) = o;
}

// --- fused GEMM + subLSTM epilogue ------------------------------------------
// 256 threads = 8 wave32; tile M=128 rows x 32 h-cols x 4 gates.
// Wave (wm 0..3, wn 0..1): rows m0+wm*32..+31, h-cols h0+wn*16..+15, all gates,
// so the epilogue combining i/f/z/o is wave-local.
__global__ __launch_bounds__(256)
void sublstm_fused_kernel(const unsigned short* __restrict__ Xb,
                          const unsigned short* __restrict__ Wb,
                          const float* __restrict__ bias,
                          const float* __restrict__ old_cell,
                          float* __restrict__ out_h,
                          float* __restrict__ out_c) {
    __shared__ __align__(16) unsigned short lA[2][128 * LDA];   // 20 KB, 2 buffers

    const int tid  = threadIdx.x;
    const int lane = tid & 31;
    const int wave = tid >> 5;
    const int wm   = wave >> 1;          // 0..3 : M sub-tile
    const int wn   = wave & 1;           // 0..1 : h sub-tile
    const int ln   = lane & 15;          // N index within 16x16 tile
    const int lh   = lane >> 4;          // half-wave selector
    const int m0   = blockIdx.y * 128;
    const int h0   = blockIdx.x * 32 + wn * 16;

    v8f acc[2][4];
    #pragma unroll
    for (int mt = 0; mt < 2; ++mt)
        #pragma unroll
        for (int g = 0; g < 4; ++g) { v8f z = {}; acc[mt][g] = z; }

    const unsigned short* atile = Xb + (long long)m0 * Kn;

#if SUBLSTM_HAVE_TDM
    const bool tdm_wave = (wave == 0);
    const unsigned lds0 = (unsigned)(unsigned long long)&lA[0][0];
    const unsigned lds1 = (unsigned)(unsigned long long)&lA[1][0];
    if (tdm_wave) tdm_load_tile_a(atile, lds0);        // preload slab 0 -> buf 0
#endif

    for (int ks = 0; ks < NSTEP; ++ks) {
        const int k0  = ks * 32;
        const int buf = ks & 1;

#if SUBLSTM_HAVE_TDM
        if (tdm_wave) {
            if (ks + 1 < NSTEP) {
                // next slab into the other buffer (its readers finished at the
                // end-of-iteration barrier of ks-1), then wait for slab ks.
                tdm_load_tile_a(atile + k0 + 32, buf ? lds0 : lds1);
                __builtin_amdgcn_s_wait_tensorcnt((short)1);
            } else {
                __builtin_amdgcn_s_wait_tensorcnt((short)0);
            }
        }
#else
        // fallback: cooperative staging, loads decoupled from stores
        {
            int c0 = tid, c1 = tid + 256;
            uint4 d0 = *(const uint4*)(atile + (long long)(c0 >> 2) * Kn + k0 + (c0 & 3) * 8);
            uint4 d1 = *(const uint4*)(atile + (long long)(c1 >> 2) * Kn + k0 + (c1 & 3) * 8);
            *(uint4*)&lA[buf][(c0 >> 2) * LDA + (c0 & 3) * 8] = d0;
            *(uint4*)&lA[buf][(c1 >> 2) * LDA + (c1 & 3) * 8] = d1;
        }
#endif
        __syncthreads();   // publish buf to all waves

        if (k0 + 32 < Kn) {
            // keep next W slab warm (emits global_prefetch_b8)
            __builtin_prefetch(Wb + (long long)(h0 + ln) * Kn + k0 + 32, 0, 3);
        }

        // ---- B fragments: column N = W row (gate g, h-col); K contiguous ----
        v16bf bfrag[4];
        #pragma unroll
        for (int g = 0; g < 4; ++g) {
            const int n = g * 256 + h0 + ln;
            const unsigned short* p = Wb + (long long)n * Kn + k0 + lh * 16;
            u16x8 blo = *(const u16x8*)p;
            u16x8 bhi = *(const u16x8*)(p + 8);
            bfrag[g] = mk_frag(blo, bhi);
        }

        // ---- A fragments from LDS + 8 WMMAs ---------------------------------
        #pragma unroll
        for (int mt = 0; mt < 2; ++mt) {
            const int row = wm * 32 + mt * 16 + ln;
            const int kb  = lh * 16;
            u16x8 alo = *(const u16x8*)&lA[buf][row * LDA + kb];
            u16x8 ahi = *(const u16x8*)&lA[buf][row * LDA + kb + 8];
            v16bf afrag = mk_frag(alo, ahi);
            #pragma unroll
            for (int g = 0; g < 4; ++g) {
                acc[mt][g] = __builtin_amdgcn_wmma_f32_16x16x32_bf16(
                    false, afrag, false, bfrag[g],
                    (short)0, acc[mt][g], false, false);
            }
        }
        __syncthreads();   // all reads of buf done; TDM may overwrite next iter
    }

    // ---- fused subLSTM epilogue (all 4 gates wave-local) --------------------
    const int h = h0 + ln;
    float bi[4];
    #pragma unroll
    for (int g = 0; g < 4; ++g) bi[g] = bias[g * 256 + h];

    #pragma unroll
    for (int mt = 0; mt < 2; ++mt) {
        #pragma unroll
        for (int r = 0; r < 8; ++r) {
            // C/D layout: VGPR r, lanes 0-15 -> M=r, lanes 16-31 -> M=r+8
            int m = m0 + wm * 32 + mt * 16 + r + lh * 8;
            float ig = sigmoidf(acc[mt][0][r] + bi[0]);
            float fg = sigmoidf(acc[mt][1][r] + bi[1]);
            float zg = sigmoidf(acc[mt][2][r] + bi[2]);
            float og = sigmoidf(acc[mt][3][r] + bi[3]);
            float oc = old_cell[(long long)m * Hn + h];
            float nc = oc * fg + zg - ig;
            float nh = sigmoidf(nc) - og;
            out_h[(long long)m * Hn + h] = nh;
            out_c[(long long)m * Hn + h] = nc;
        }
    }
}

extern "C" void kernel_launch(void* const* d_in, const int* in_sizes, int n_in,
                              void* d_out, int out_size, void* d_ws, size_t ws_size,
                              hipStream_t stream) {
    const float* input    = (const float*)d_in[0];   // [B, 256]
    const float* weights  = (const float*)d_in[1];   // [1024, 512]
    const float* bias     = (const float*)d_in[2];   // [1024]
    const float* old_h    = (const float*)d_in[3];   // [B, 256]
    const float* old_cell = (const float*)d_in[4];   // [B, 256]

    float* out_h = (float*)d_out;                    // [B, 256]
    float* out_c = out_h + (long long)Bn * Hn;       // [B, 256]

    unsigned short* Xb = (unsigned short*)d_ws;              // bf16 [B,512]
    unsigned short* Wb = Xb + (size_t)Bn * Kn;               // bf16 [1024,512]

    {
        long long quads = (long long)Bn * Kn / 4;
        int blocks = (int)((quads + 255) / 256);
        pack_x_kernel<<<blocks, 256, 0, stream>>>(input, old_h, Xb);
    }
    {
        long long quads = (long long)1024 * Kn / 4;
        int blocks = (int)((quads + 255) / 256);
        pack_w_kernel<<<blocks, 256, 0, stream>>>(weights, Wb);
    }
    {
        dim3 grid(Hn / 32, Bn / 128);   // (8, 256)
        sublstm_fused_kernel<<<grid, 256, 0, stream>>>(Xb, Wb, bias, old_cell,
                                                       out_h, out_c);
    }
}
